// ATLSemanticHubV6_89644557402932
// MI455X (gfx1250) — compile-verified, matrix operand order checked by reading, not run on hardware
//
#include <hip/hip_runtime.h>

typedef __attribute__((ext_vector_type(16))) __bf16 v16bf;
typedef __attribute__((ext_vector_type(8)))  __bf16 v8bf;
typedef __attribute__((ext_vector_type(8)))  float  v8f;

#define M_TILE  32
#define TPR     8           // threads per row in phases 1/3/4 (256 threads / 32 rows)
#define D_DIM   256
#define P_DIM   512
#define KSEL    5
#define SIMS_LD 516         // 512 + 4 pad -> row r offset by 4r banks
#define A_LD    264         // 256 + 8 bf16 pad -> 16B-aligned rows, bank-staggered

union FragBF {
    v16bf v;
    v8bf  h[2];
};

__device__ __forceinline__ void top5_insert(float v, int c,
        float& b0, float& b1, float& b2, float& b3, float& b4,
        int& i0, int& i1, int& i2, int& i3, int& i4) {
    if (v <= b4) return;
    if (v > b0)      { b4=b3;i4=i3; b3=b2;i3=i2; b2=b1;i2=i1; b1=b0;i1=i0; b0=v;i0=c; }
    else if (v > b1) { b4=b3;i4=i3; b3=b2;i3=i2; b2=b1;i2=i1; b1=v; i1=c; }
    else if (v > b2) { b4=b3;i4=i3; b3=b2;i3=i2; b2=v; i2=c; }
    else if (v > b3) { b4=b3;i4=i3; b3=v; i3=c; }
    else             { b4=v; i4=c; }
}

// ---- prep: fp32 prototypes -> bf16 (row-major, 512x256), runs once, ~256KB ----
__global__ __launch_bounds__(256)
void proto_bf16_kernel(const float* __restrict__ p, __bf16* __restrict__ pb) {
    const int i = blockIdx.x * 256 + threadIdx.x;      // 8 floats per thread
    const float4* src = reinterpret_cast<const float4*>(p) + i * 2;
    float4 a = src[0], b = src[1];
    v8bf o;
    o[0]=(__bf16)a.x; o[1]=(__bf16)a.y; o[2]=(__bf16)a.z; o[3]=(__bf16)a.w;
    o[4]=(__bf16)b.x; o[5]=(__bf16)b.y; o[6]=(__bf16)b.z; o[7]=(__bf16)b.w;
    *reinterpret_cast<v8bf*>(pb + i * 8) = o;
}

template <bool PRECONV>
__global__ __launch_bounds__(256)
void ATLSemanticHub_kernel(const float* __restrict__ x,
                           const float* __restrict__ protosF,
                           const __bf16* __restrict__ protosB,
                           float* __restrict__ out)
{
    __shared__ __attribute__((aligned(16))) __bf16 Alds[M_TILE * A_LD];
    __shared__ float sims[M_TILE * SIMS_LD];
    __shared__ float candV[M_TILE * TPR * KSEL];
    __shared__ int   candI[M_TILE * TPR * KSEL];
    __shared__ float finW[M_TILE * KSEL];
    __shared__ int   finI[M_TILE * KSEL];

    const int tid = threadIdx.x;
    const int r   = tid >> 3;          // local row 0..31
    const int t   = tid & 7;           // 8 threads per row
    const long gRow = (long)blockIdx.x * M_TILE + r;

    // ---------------- Phase 1: row norms + bf16 A tile in LDS ----------------
    const float4* xrow = reinterpret_cast<const float4*>(x + gRow * D_DIM) + t * 8;
    float4 f[8];
    #pragma unroll
    for (int i = 0; i < 8; ++i) f[i] = xrow[i];

    float ss = 0.0f;
    #pragma unroll
    for (int i = 0; i < 8; ++i)
        ss += f[i].x*f[i].x + f[i].y*f[i].y + f[i].z*f[i].z + f[i].w*f[i].w;
    // butterfly across the 8 lanes of this row (xor bits 0..2 stay in-group)
    #pragma unroll
    for (int o = 4; o > 0; o >>= 1) ss += __shfl_xor(ss, o, 32);
    const float inv = 1.0f / fmaxf(sqrtf(ss), 1e-12f);

    #pragma unroll
    for (int i = 0; i < 8; i += 2) {
        v8bf o;
        o[0]=(__bf16)(f[i].x*inv);   o[1]=(__bf16)(f[i].y*inv);
        o[2]=(__bf16)(f[i].z*inv);   o[3]=(__bf16)(f[i].w*inv);
        o[4]=(__bf16)(f[i+1].x*inv); o[5]=(__bf16)(f[i+1].y*inv);
        o[6]=(__bf16)(f[i+1].z*inv); o[7]=(__bf16)(f[i+1].w*inv);
        *reinterpret_cast<v8bf*>(&Alds[r * A_LD + t * 32 + i * 4]) = o;
    }
    __syncthreads();

    // ---------------- Phase 2: WMMA sims = xn @ protos^T ----------------
    const int lane = tid & 31;
    const int w    = tid >> 5;                 // wave 0..7
    const int mrow = lane & 15;
    const int koff = (lane < 16) ? 0 : 8;      // ISA 16-bit A/B fragment layout

    FragBF afrag0[8], afrag1[8];               // row-tiles [0..15] and [16..31]
    #pragma unroll
    for (int ks = 0; ks < 8; ++ks) {
        const __bf16* a0 = &Alds[mrow * A_LD + ks * 32 + koff];
        const __bf16* a1 = &Alds[(mrow + 16) * A_LD + ks * 32 + koff];
        afrag0[ks].h[0] = *reinterpret_cast<const v8bf*>(a0);
        afrag0[ks].h[1] = *reinterpret_cast<const v8bf*>(a0 + 16);
        afrag1[ks].h[0] = *reinterpret_cast<const v8bf*>(a1);
        afrag1[ks].h[1] = *reinterpret_cast<const v8bf*>(a1 + 16);
    }

    #pragma unroll
    for (int j = 0; j < 4; ++j) {
        const int n0 = (w * 4 + j) * 16;
        const int n  = n0 + mrow;
        v8f acc0 = {}, acc1 = {};
        #pragma unroll
        for (int ks = 0; ks < 8; ++ks) {
            const int c = ks * 32 + koff;
            FragBF bf_;
            if constexpr (PRECONV) {
                const __bf16* prow = protosB + n * D_DIM;
                bf_.h[0] = *reinterpret_cast<const v8bf*>(prow + c);
                bf_.h[1] = *reinterpret_cast<const v8bf*>(prow + c + 16);
            } else {
                const float* prow = protosF + n * D_DIM;
                float4 b0 = *reinterpret_cast<const float4*>(prow + c);
                float4 b1 = *reinterpret_cast<const float4*>(prow + c + 4);
                float4 b2 = *reinterpret_cast<const float4*>(prow + c + 16);
                float4 b3 = *reinterpret_cast<const float4*>(prow + c + 20);
                bf_.h[0][0]=(__bf16)b0.x; bf_.h[0][1]=(__bf16)b0.y; bf_.h[0][2]=(__bf16)b0.z; bf_.h[0][3]=(__bf16)b0.w;
                bf_.h[0][4]=(__bf16)b1.x; bf_.h[0][5]=(__bf16)b1.y; bf_.h[0][6]=(__bf16)b1.z; bf_.h[0][7]=(__bf16)b1.w;
                bf_.h[1][0]=(__bf16)b2.x; bf_.h[1][1]=(__bf16)b2.y; bf_.h[1][2]=(__bf16)b2.z; bf_.h[1][3]=(__bf16)b2.w;
                bf_.h[1][4]=(__bf16)b3.x; bf_.h[1][5]=(__bf16)b3.y; bf_.h[1][6]=(__bf16)b3.z; bf_.h[1][7]=(__bf16)b3.w;
            }
            acc0 = __builtin_amdgcn_wmma_f32_16x16x32_bf16(
                false, afrag0[ks].v, false, bf_.v, (short)0, acc0, false, false);
            acc1 = __builtin_amdgcn_wmma_f32_16x16x32_bf16(
                false, afrag1[ks].v, false, bf_.v, (short)0, acc1, false, false);
        }
        // C/D layout: lane<16 -> (M=i, N=lane); lane>=16 -> (M=i+8, N=lane-16)
        #pragma unroll
        for (int i = 0; i < 8; ++i) {
            const int m = (lane < 16) ? i : (i + 8);
            sims[m * SIMS_LD + n]        = acc0[i];
            sims[(m + 16) * SIMS_LD + n] = acc1[i];
        }
    }
    __syncthreads();

    // ---------------- Phase 3: two-level top-5 + softmax ----------------
    {
        float b0=-3.0e38f,b1=-3.0e38f,b2=-3.0e38f,b3=-3.0e38f,b4=-3.0e38f;
        int   i0=0,i1=0,i2=0,i3=0,i4=0;
        #pragma unroll 4
        for (int jj = 0; jj < 64; ++jj) {
            const int c = t + (jj << 3);
            top5_insert(sims[r * SIMS_LD + c], c, b0,b1,b2,b3,b4, i0,i1,i2,i3,i4);
        }
        const int cb = (r * TPR + t) * KSEL;
        candV[cb+0]=b0; candV[cb+1]=b1; candV[cb+2]=b2; candV[cb+3]=b3; candV[cb+4]=b4;
        candI[cb+0]=i0; candI[cb+1]=i1; candI[cb+2]=i2; candI[cb+3]=i3; candI[cb+4]=i4;
    }
    __syncthreads();

    if (tid < M_TILE) {
        const int row  = tid;
        const int base = row * TPR * KSEL;
        float b0=-3.0e38f,b1=-3.0e38f,b2=-3.0e38f,b3=-3.0e38f,b4=-3.0e38f;
        int   i0=0,i1=0,i2=0,i3=0,i4=0;
        for (int q = 0; q < TPR * KSEL; ++q)
            top5_insert(candV[base + q], candI[base + q], b0,b1,b2,b3,b4, i0,i1,i2,i3,i4);
        // temperature softmax over the 5 winners (b0 is the max; 1/T = 5)
        const float e0 = 1.0f;
        const float e1 = __expf((b1 - b0) * 5.0f);
        const float e2 = __expf((b2 - b0) * 5.0f);
        const float e3 = __expf((b3 - b0) * 5.0f);
        const float e4 = __expf((b4 - b0) * 5.0f);
        const float rs = 1.0f / (e0 + e1 + e2 + e3 + e4);
        const int fb = row * KSEL;
        finI[fb+0]=i0; finI[fb+1]=i1; finI[fb+2]=i2; finI[fb+3]=i3; finI[fb+4]=i4;
        finW[fb+0]=e0*rs; finW[fb+1]=e1*rs; finW[fb+2]=e2*rs; finW[fb+3]=e3*rs; finW[fb+4]=e4*rs;
    }
    __syncthreads();

    // ---------------- Phase 4: sparse scatter into dense output ----------------
    // Thread owns out[gRow, t*64 .. t*64+63]: zero-fill then patch its winners.
    // Same-thread store->store to the same address stays ordered within a wave.
    float* orow = out + gRow * P_DIM + t * 64;
    const float4 z = {0.0f, 0.0f, 0.0f, 0.0f};
    #pragma unroll
    for (int i = 0; i < 16; ++i)
        reinterpret_cast<float4*>(orow)[i] = z;
    const int fb = r * KSEL;
    #pragma unroll
    for (int q = 0; q < KSEL; ++q) {
        const int d = finI[fb + q] - t * 64;
        if ((unsigned)d < 64u) orow[d] = finW[fb + q];
    }
}

extern "C" void kernel_launch(void* const* d_in, const int* in_sizes, int n_in,
                              void* d_out, int out_size, void* d_ws, size_t ws_size,
                              hipStream_t stream) {
    const float* x      = (const float*)d_in[0];
    const float* protos = (const float*)d_in[1];
    float*       out    = (float*)d_out;

    const int B    = in_sizes[0] / D_DIM;    // 262144
    const int grid = B / M_TILE;             // 8192 blocks

    const size_t needBf = (size_t)P_DIM * D_DIM * sizeof(__bf16);  // 256 KB
    if (ws_size >= needBf) {
        __bf16* pb = (__bf16*)d_ws;
        proto_bf16_kernel<<<dim3((P_DIM * D_DIM / 8) / 256), dim3(256), 0, stream>>>(protos, pb);
        ATLSemanticHub_kernel<true><<<dim3(grid), dim3(256), 0, stream>>>(x, protos, pb, out);
    } else {
        ATLSemanticHub_kernel<false><<<dim3(grid), dim3(256), 0, stream>>>(x, protos, nullptr, out);
    }
}
